// PatchAttn_17136919511598
// MI455X (gfx1250) — matrix operand order repaired and checked by reading, not compiled, not run on hardware
//
#include <hip/hip_runtime.h>

// ---------------------------------------------------------------------------
// PatchAttn on MI455X (gfx1250, wave32, WMMA + Tensor Data Mover).
//
// y = space_to_depth(x,4): (8,1024,64,64); Q = y flat (8,1024,4096)
// energy = Q Q^T (1024x1024x4096)   -> TDM tiles -> LDS -> v_wmma bf16
// attn   = softmax(rowmax - energy) == exp(rowmin - e)/sum   (f32)
// out    = attn Q (1024x4096x1024)  -> TDM tiles -> LDS -> v_wmma bf16
// final  = gamma*out + y, scattered back through depth_to_space (bijection).
//
// GEMM blocks: 128x128 per workgroup (4 waves, one 64x64 quadrant each).
// Per k-step (K=32): wave0 DMAs A(128x32) + B(128x32) bf16 tiles into LDS
// (double buffered, TENSORcnt), all waves consume via ds_load_b128.
// ---------------------------------------------------------------------------

typedef __attribute__((ext_vector_type(16))) __bf16 v16bf;
typedef __attribute__((ext_vector_type(8)))  float  v8f;
typedef unsigned int uint32x4 __attribute__((ext_vector_type(4)));
typedef int          int32x8  __attribute__((ext_vector_type(8)));
typedef int          int32x4  __attribute__((ext_vector_type(4)));

#define NB   8      // batches
#define C2   1024   // channels after space_to_depth
#define NSP  4096   // 64*64 spatial
#define CH   64     // original channels
#define HW   256    // original H=W

#define TILE_BYTES   8192u            // 128 rows * 32 bf16 * 2B
#define BUF_BYTES    (2u * TILE_BYTES)  // A tile + B tile
#define SMEM_BYTES   (2u * BUF_BYTES)   // double buffered

__device__ __forceinline__ unsigned short f2bf(float f) {
  union { float f; unsigned u; } v; v.f = f;
  unsigned r = v.u + 0x7FFFu + ((v.u >> 16) & 1u);   // RNE
  return (unsigned short)(r >> 16);
}

// --------- Tensor Data Mover: 2D tile (rows x 32 bf16) global -> LDS --------
// D# per CDNA5 ISA ch.8: group0 = {count=1, lds_addr, global_addr, type=2},
// group1 = {data_size=1(2B), tensor dims (large, never clips), tile 32 x rows,
//           tensor_dim0_stride = row stride in elements}.
__device__ __forceinline__ void tdm_load_tile(unsigned lds_addr, const void* gptr,
                                              unsigned rows, unsigned stride_elems) {
  union { uint32x4 v; unsigned long long q[2]; } g0;
  unsigned long long ga = (unsigned long long)(size_t)gptr;
  g0.q[0] = 1ull | ((unsigned long long)lds_addr << 32);          // count=1 | lds
  g0.q[1] = (ga & 0x01FFFFFFFFFFFFFFull) | (2ull << 62);          // addr | type=2

  union { int32x8 v; unsigned long long q[4]; } g1;
  const unsigned long long TD0 = 1ull << 20;                      // tensor dims:
  const unsigned long long TD1 = 1ull << 20;                      // big, no clip
  g1.q[0] = (1ull << 16)                                          // data_size=2B
          | ((TD0 & 0xFFFFull) << 48);                            // dim0[15:0]
  g1.q[1] = ((TD0 >> 16) & 0xFFFFull)                             // dim0[31:16]
          | ((TD1 & 0xFFFFFFFFull) << 16)                         // dim1
          | (32ull << 48);                                        // tile_dim0=32
  g1.q[2] = (unsigned long long)rows                              // tile_dim1
          | ((unsigned long long)stride_elems << 32);             // d0 stride lo
  g1.q[3] = 0;                                                    // stride hi etc

  int32x4 z4 = {};
#if __clang_major__ >= 23
  int32x8 z8 = {};
  __builtin_amdgcn_tensor_load_to_lds(g0.v, g1.v, z4, z4, z8, 0);
#else
  __builtin_amdgcn_tensor_load_to_lds(g0.v, g1.v, z4, z4, 0);
#endif
}

// --------- WMMA fragment gathers from a staged LDS tile (rows of 32 bf16) ---
// A 16x32 bf16 (ISA 7.12.2): lanes 0-15 K={0..7,16..23}, lanes 16-31 K={8..15,24..31}
__device__ __forceinline__ v16bf lds_a_frag(const char* tile, int row, int half) {
  union { v16bf v; uint4 u[2]; } f;
  const char* p = tile + row * 64 + half * 16;
  f.u[0] = *reinterpret_cast<const uint4*>(p);
  f.u[1] = *reinterpret_cast<const uint4*>(p + 32);
  return f.v;
}
// B 32x16 bf16: column N=lane&15; lanes 0-15 K={0..7,16..23}, lanes 16-31 K={8..15,24..31}
__device__ __forceinline__ v16bf lds_b_frag(const char* tile, int row, int half) {
  union { v16bf v; uint4 u[2]; } f;
  const char* p = tile + row * 64 + half * 16;
  f.u[0] = *reinterpret_cast<const uint4*>(p);
  f.u[1] = *reinterpret_cast<const uint4*>(p + 32);
  return f.v;
}

// ------------------------- pack: x -> Q (bf16) and Q^T (bf16) ----------------
__global__ void __launch_bounds__(256)
pack_kernel(const float* __restrict__ x,
            unsigned short* __restrict__ Q,
            unsigned short* __restrict__ QT) {
  size_t t = (size_t)blockIdx.x * 256 + threadIdx.x;   // over NB*C2*NSP = 2^25
  int n  = (int)(t & (NSP - 1));
  int c2 = (int)((t >> 12) & (C2 - 1));
  int b  = (int)(t >> 22);
  int c  = c2 & 63, ij = c2 >> 6, ii = ij >> 2, jj = ij & 3;
  int hb = n >> 6,  wb = n & 63;
  size_t addr = (((size_t)b * CH + c) * HW + (hb * 4 + ii)) * HW + (wb * 4 + jj);
  unsigned short us = f2bf(x[addr]);
  Q[t] = us;                                            // [b][c2][n]
  QT[((size_t)b * NSP + n) * C2 + c2] = us;             // [b][n][c2]
}

// ------------------------- energy = Q Q^T ------------------------------------
// 4 waves per block; 128x128 output block; K = 4096; TDM double buffering.
__global__ void __launch_bounds__(128)
gram_kernel(const unsigned short* __restrict__ Q, float* __restrict__ E) {
  __shared__ __align__(128) char smem[SMEM_BYTES];
  const int tid  = threadIdx.x;
  const int lane = tid & 31, l16 = lane & 15, half = (lane >> 4) & 1;
  const int wave = tid >> 5, wi = wave >> 1, wj = wave & 1;
  const int nBase = blockIdx.x * 128;
  const int mBase = blockIdx.y * 128;
  const int b     = blockIdx.z;
  const unsigned short* Qb    = Q + (size_t)b * C2 * NSP;
  const unsigned short* Abase = Qb + (size_t)mBase * NSP;
  const unsigned short* Bbase = Qb + (size_t)nBase * NSP;
  const unsigned ldsbase = (unsigned)(size_t)&smem[0];   // LDS byte offset

  v8f acc[4][4];
#pragma unroll
  for (int i = 0; i < 4; ++i)
#pragma unroll
    for (int j = 0; j < 4; ++j) acc[i][j] = {};

  if (wave == 0) {                                       // preload buffer 0
    tdm_load_tile(ldsbase,              Abase, 128, NSP);
    tdm_load_tile(ldsbase + TILE_BYTES, Bbase, 128, NSP);
  }

  const int ITER = NSP / 32;                             // 128
  for (int it = 0; it < ITER; ++it) {
    const int cur = it & 1;
    if (wave == 0) {
      if (it + 1 < ITER) {
        const int kb = (it + 1) * 32;
        const unsigned boff = ldsbase + (cur ^ 1) * BUF_BYTES;
        tdm_load_tile(boff,              Abase + kb, 128, NSP);
        tdm_load_tile(boff + TILE_BYTES, Bbase + kb, 128, NSP);
        __builtin_amdgcn_s_wait_tensorcnt(2);            // cur's 2 DMAs done
      } else {
        __builtin_amdgcn_s_wait_tensorcnt(0);
      }
    }
    __syncthreads();                                     // publish buf[cur]

    const char* As = smem + cur * BUF_BYTES;
    const char* Bs = As + TILE_BYTES;
    v16bf a[4], bm[4];
#pragma unroll
    for (int i = 0; i < 4; ++i) a[i]  = lds_a_frag(As, wi * 64 + i * 16 + l16, half);
#pragma unroll
    for (int j = 0; j < 4; ++j) bm[j] = lds_b_frag(Bs, wj * 64 + j * 16 + l16, half);
#pragma unroll
    for (int i = 0; i < 4; ++i)
#pragma unroll
      for (int j = 0; j < 4; ++j)
        acc[i][j] = __builtin_amdgcn_wmma_f32_16x16x32_bf16(
            false, a[i], false, bm[j], (short)0, acc[i][j], false, false);

    __syncthreads();                                     // done reading buf[cur]
  }

  float* Eb = E + (size_t)b * C2 * C2;
#pragma unroll
  for (int i = 0; i < 4; ++i)
#pragma unroll
    for (int j = 0; j < 4; ++j) {
      union { v8f v; float s[8]; } u; u.v = acc[i][j];
#pragma unroll
      for (int r = 0; r < 8; ++r) {
        int m = mBase + wi * 64 + i * 16 + r + 8 * half; // C/D layout
        int n = nBase + wj * 64 + j * 16 + l16;
        Eb[(size_t)m * C2 + n] = u.s[r];
      }
    }
}

// ------------------------- attn = softmax(rowmax - E) = exp(rowmin - e)/sum --
__global__ void __launch_bounds__(256)
softmax_kernel(const float* __restrict__ E, unsigned short* __restrict__ A) {
  const int rowg = blockIdx.x;                           // b*1024 + c2
  const float* e = E + (size_t)rowg * C2;
  unsigned short* a = A + (size_t)rowg * C2;
  __shared__ float red[256];
  const int t = threadIdx.x;
  float v0 = e[t], v1 = e[t + 256], v2 = e[t + 512], v3 = e[t + 768];
  float mn = fminf(fminf(v0, v1), fminf(v2, v3));
  red[t] = mn; __syncthreads();
  for (int s = 128; s > 0; s >>= 1) { if (t < s) red[t] = fminf(red[t], red[t + s]); __syncthreads(); }
  mn = red[0]; __syncthreads();
  float t0 = expf(mn - v0), t1 = expf(mn - v1), t2 = expf(mn - v2), t3 = expf(mn - v3);
  red[t] = t0 + t1 + t2 + t3; __syncthreads();
  for (int s = 128; s > 0; s >>= 1) { if (t < s) red[t] += red[t + s]; __syncthreads(); }
  float inv = 1.0f / red[0];
  a[t]       = f2bf(t0 * inv);
  a[t + 256] = f2bf(t1 * inv);
  a[t + 512] = f2bf(t2 * inv);
  a[t + 768] = f2bf(t3 * inv);
}

// ------------------------- out = gamma*(attn Q) + y, depth_to_space ----------
__global__ void __launch_bounds__(128)
attnv_kernel(const unsigned short* __restrict__ A,
             const unsigned short* __restrict__ QT,
             const float* __restrict__ x,
             const float* __restrict__ gamma,
             float* __restrict__ out) {
  __shared__ __align__(128) char smem[SMEM_BYTES];
  const int tid  = threadIdx.x;
  const int lane = tid & 31, l16 = lane & 15, half = (lane >> 4) & 1;
  const int wave = tid >> 5, wi = wave >> 1, wj = wave & 1;
  const int nBase = blockIdx.x * 128;                    // spatial
  const int mBase = blockIdx.y * 128;                    // c2
  const int b     = blockIdx.z;
  const unsigned short* Abase = A  + (size_t)b * C2 * C2  + (size_t)mBase * C2;
  const unsigned short* Bbase = QT + (size_t)b * NSP * C2 + (size_t)nBase * C2;
  const unsigned ldsbase = (unsigned)(size_t)&smem[0];

  v8f acc[4][4];
#pragma unroll
  for (int i = 0; i < 4; ++i)
#pragma unroll
    for (int j = 0; j < 4; ++j) acc[i][j] = {};

  if (wave == 0) {
    tdm_load_tile(ldsbase,              Abase, 128, C2);
    tdm_load_tile(ldsbase + TILE_BYTES, Bbase, 128, C2);
  }

  const int ITER = C2 / 32;                              // 32
  for (int it = 0; it < ITER; ++it) {
    const int cur = it & 1;
    if (wave == 0) {
      if (it + 1 < ITER) {
        const int kb = (it + 1) * 32;
        const unsigned boff = ldsbase + (cur ^ 1) * BUF_BYTES;
        tdm_load_tile(boff,              Abase + kb, 128, C2);
        tdm_load_tile(boff + TILE_BYTES, Bbase + kb, 128, C2);
        __builtin_amdgcn_s_wait_tensorcnt(2);
      } else {
        __builtin_amdgcn_s_wait_tensorcnt(0);
      }
    }
    __syncthreads();

    const char* As = smem + cur * BUF_BYTES;
    const char* Bs = As + TILE_BYTES;
    v16bf a[4], bm[4];
#pragma unroll
    for (int i = 0; i < 4; ++i) a[i]  = lds_a_frag(As, wi * 64 + i * 16 + l16, half);
#pragma unroll
    for (int j = 0; j < 4; ++j) bm[j] = lds_b_frag(Bs, wj * 64 + j * 16 + l16, half);
#pragma unroll
    for (int i = 0; i < 4; ++i)
#pragma unroll
      for (int j = 0; j < 4; ++j)
        acc[i][j] = __builtin_amdgcn_wmma_f32_16x16x32_bf16(
            false, a[i], false, bm[j], (short)0, acc[i][j], false, false);

    __syncthreads();
  }

  const float g = gamma[0];
#pragma unroll
  for (int i = 0; i < 4; ++i)
#pragma unroll
    for (int j = 0; j < 4; ++j) {
      union { v8f v; float s[8]; } u; u.v = acc[i][j];
#pragma unroll
      for (int r = 0; r < 8; ++r) {
        int m = mBase + wi * 64 + i * 16 + r + 8 * half; // c2
        int n = nBase + wj * 64 + j * 16 + l16;          // spatial
        int c = m & 63, ij = m >> 6, ii = ij >> 2, jj = ij & 3;
        int hb = n >> 6, wb = n & 63;
        size_t addr = (((size_t)b * CH + c) * HW + (hb * 4 + ii)) * HW + (wb * 4 + jj);
        out[addr] = g * u.s[r] + x[addr];                // residual y == x[addr]
      }
    }
}

extern "C" void kernel_launch(void* const* d_in, const int* in_sizes, int n_in,
                              void* d_out, int out_size, void* d_ws, size_t ws_size,
                              hipStream_t stream) {
  const float* x     = (const float*)d_in[0];
  const float* gamma = (const float*)d_in[1];
  float* out = (float*)d_out;

  char* ws = (char*)d_ws;
  unsigned short* Q    = (unsigned short*)(ws);                          //  64 MiB
  unsigned short* QT   = (unsigned short*)(ws + ((size_t)64  << 20));    //  64 MiB
  float*          E    = (float*)        (ws + ((size_t)128 << 20));     //  32 MiB
  unsigned short* Attn = (unsigned short*)(ws + ((size_t)160 << 20));    //  16 MiB

  pack_kernel<<<(NB * C2 * NSP) / 256, 256, 0, stream>>>(x, Q, QT);

  dim3 g1(C2 / 128, C2 / 128, NB);          // (8,8,8)
  gram_kernel<<<g1, 128, 0, stream>>>(Q, E);

  softmax_kernel<<<NB * C2, 256, 0, stream>>>(E, Attn);

  dim3 g2(NSP / 128, C2 / 128, NB);         // (32,8,8)
  attnv_kernel<<<g2, 128, 0, stream>>>(Attn, QT, x, gamma, out);
}
